// MoleculeEncoder_6820408066372
// MI455X (gfx1250) — matrix operand order repaired and checked by reading, not compiled, not run on hardware
//
#include <hip/hip_runtime.h>
#include <hip/hip_bf16.h>

// ---------------------------------------------------------------------------
// CDNA5 (gfx1250) molecule encoder: bf16 WMMA GEMMs + vectorized gathers.
//  - weights pre-packed to bf16, transposed [N][Kpad]
//  - B tiles staged by the Tensor Data Mover (tensor_load_to_lds) when the
//    builtin exists; fallback: unconditional global b128 -> ds b128
//  - A tiles: float4-vectorized fp32->bf16 staging when lda is 4-aligned
// ---------------------------------------------------------------------------

typedef __attribute__((ext_vector_type(16))) __bf16 v16bf;
typedef __attribute__((ext_vector_type(8)))  float  v8f;
typedef __attribute__((ext_vector_type(4)))  unsigned int u32x4;
typedef __attribute__((ext_vector_type(8)))  int          i32x8;
typedef __attribute__((ext_vector_type(4)))  int          i32x4;

#if __has_builtin(__builtin_amdgcn_tensor_load_to_lds) && \
    __has_builtin(__builtin_amdgcn_s_wait_tensorcnt)
#define HAVE_TDM 1
#else
#define HAVE_TDM 0
#endif

union FragBF { v16bf v; unsigned short u[16]; };

__device__ __forceinline__ unsigned short f2bf(float f) {
    unsigned int u = __float_as_uint(f);
    u += 0x7FFFu + ((u >> 16) & 1u);     // round-to-nearest-even
    return (unsigned short)(u >> 16);
}
__device__ __forceinline__ unsigned int pk2bf(float a, float b) {
    return (unsigned int)f2bf(a) | ((unsigned int)f2bf(b) << 16);
}

#define BM 128
#define BN 128
#define BK 32
#define LDP 40   // padded LDS K-stride: 80 B row stride (16B aligned, conflict-free)

// Pack W[K,N] (fp32, row-major [in,out]) -> Wt[N][Kpad] bf16, zero-padded K.
__global__ void __launch_bounds__(256)
pack_wt(const float* __restrict__ W, unsigned short* __restrict__ Wt,
        int K, int N, int Kpad)
{
    int n = blockIdx.y;
    int k = blockIdx.x * 256 + threadIdx.x;
    if (k >= Kpad) return;
    float v = (k < K) ? W[(size_t)k * N + n] : 0.0f;
    Wt[(size_t)n * Kpad + k] = f2bf(v);
}

// C[M,N] = act( Base + [A0|A1] @ W + bias ), optionally also store pre-act.
__global__ void __launch_bounds__(256)
gemm_bf16_wmma(const float* __restrict__ A0, int lda0, int K0,
               const float* __restrict__ A1, int lda1, int K1,
               const unsigned short* __restrict__ Wt, int Kpad,
               const float* __restrict__ Base,
               const float* __restrict__ bias,
               float* __restrict__ Cpre,
               float* __restrict__ C,
               int M, int N, int doRelu)
{
    // Single LDS allocation so the Bs byte offset is known for the TDM D#.
    __shared__ __align__(16) unsigned short smem[BM * LDP + BN * LDP];
    unsigned short (*As)[LDP] = reinterpret_cast<unsigned short(*)[LDP]>(&smem[0]);
    unsigned short (*Bs)[LDP] = reinterpret_cast<unsigned short(*)[LDP]>(&smem[BM * LDP]);
    const unsigned int BS_LDS_BYTE_OFF = BM * LDP * 2;   // 10240

    const int t    = threadIdx.x;
    const int lane = t & 31;
    const int wave = t >> 5;     // 8 waves
    const int wm   = wave >> 2;  // 0..1 -> 64-row slab
    const int wn   = wave & 3;   // 0..3 -> 32-col slab
    const int lo   = lane & 15;
    const int hi   = lane >> 4;

    const int rowTile = blockIdx.y * BM;
    const int colTile = blockIdx.x * BN;
    const int Ktot    = K0 + K1;

    // staging ids: 2 threads per row, each handles one 16-element half (32 B)
    const int srow = t >> 1;
    const int c2   = t & 1;

    v8f acc[4][2];
    #pragma unroll
    for (int i = 0; i < 4; i++)
        #pragma unroll
        for (int j = 0; j < 2; j++)
            #pragma unroll
            for (int r = 0; r < 8; r++)
                acc[i][j][r] = 0.0f;

    for (int k0 = 0; k0 < Ktot; k0 += BK) {
        // ---- B tile: TDM async tensor load (wave 0 issues; EXEC-independent) ----
#if HAVE_TDM
        if (wave == 0) {
            unsigned long long ga =
                (unsigned long long)(const void*)(Wt + (size_t)colTile * Kpad + k0);
            u32x4 g0;
            g0[0] = 1u;                               // count=1 (valid user D#)
            g0[1] = BS_LDS_BYTE_OFF;                  // lds_addr
            g0[2] = (unsigned int)ga;                 // global_addr[31:0]
            g0[3] = (unsigned int)((ga >> 32) & 0x01FFFFFFu) | 0x80000000u; // addr[56:32] | type=2
            i32x8 g1;
            g1[0] = (1 << 16)      // data_size = 2 B
                  | (1 << 20)      // pad_enable
                  | (3 << 22)      // pad_interval: 16 DWORDs (64 B)
                  | (3 << 25);     // pad_amount:    4 DWORDs (16 B) -> 80 B row stride
            g1[1] = (Kpad & 0xFFFF) << 16;                    // tensor_dim0[15:0]
            g1[2] = ((Kpad >> 16) & 0xFFFF)                   // tensor_dim0[31:16]
                  | ((unsigned)0x0010 << 16);                 // tensor_dim1[15:0] (0x10, so dim1 = 2^20 + 16)
            g1[3] = ((1 << 20) >> 16)                         // tensor_dim1[31:16]
                  | (32 << 16);                               // tile_dim0 = 32
            g1[4] = 128;                                      // tile_dim1 = 128, tile_dim2 = 0
            g1[5] = Kpad;                                     // tensor_dim0_stride[31:0]
            g1[6] = 0;                                        // stride hi / dim1_stride lo
            g1[7] = 0;
            i32x4 g2 = (i32x4)0;
            i32x4 g3 = (i32x4)0;
            i32x8 g4 = (i32x8)0;
            __builtin_amdgcn_tensor_load_to_lds(g0, g1, g2, g3, g4, 0);
        }
#else
        {
            const unsigned short* wp =
                Wt + (size_t)(colTile + srow) * Kpad + k0 + c2 * 16;
            uint4 w0 = *reinterpret_cast<const uint4*>(wp);
            uint4 w1 = *reinterpret_cast<const uint4*>(wp + 8);
            *reinterpret_cast<uint4*>(&Bs[srow][c2 * 16 + 0]) = w0;
            *reinterpret_cast<uint4*>(&Bs[srow][c2 * 16 + 8]) = w1;
        }
#endif

        // ---- A tile: branchless fp32 -> bf16 staging ----
        const float* srcp = nullptr;
        int lds_ = 0, koff = 0;
        bool fastA = false;
        if (k0 + BK <= K0)                    { srcp = A0; lds_ = lda0; koff = k0;      fastA = true; }
        else if (k0 >= K0 && k0 + BK <= Ktot) { srcp = A1; lds_ = lda1; koff = k0 - K0; fastA = true; }
        const bool vecA = fastA && ((lds_ & 3) == 0) && ((koff & 3) == 0);

        if (vecA) {
            // float4-vectorized: 1 row, 16 consecutive k per thread
            int row = rowTile + srow;
            row = (row < M) ? row : (M - 1);
            const float* ap = srcp + (size_t)row * lds_ + koff + c2 * 16;
            float4 f0 = *reinterpret_cast<const float4*>(ap + 0);
            float4 f1 = *reinterpret_cast<const float4*>(ap + 4);
            float4 f2 = *reinterpret_cast<const float4*>(ap + 8);
            float4 f3 = *reinterpret_cast<const float4*>(ap + 12);
            uint4 p0, p1;
            p0.x = pk2bf(f0.x, f0.y); p0.y = pk2bf(f0.z, f0.w);
            p0.z = pk2bf(f1.x, f1.y); p0.w = pk2bf(f1.z, f1.w);
            p1.x = pk2bf(f2.x, f2.y); p1.y = pk2bf(f2.z, f2.w);
            p1.z = pk2bf(f3.x, f3.y); p1.w = pk2bf(f3.z, f3.w);
            *reinterpret_cast<uint4*>(&As[srow][c2 * 16 + 0]) = p0;
            *reinterpret_cast<uint4*>(&As[srow][c2 * 16 + 8]) = p1;
        } else if (fastA) {
            #pragma unroll
            for (int i = 0; i < (BM * BK) / 256; i++) {
                int linear = t + i * 256;
                int kk  = linear & (BK - 1);
                int m   = linear >> 5;
                int row = rowTile + m;
                row = (row < M) ? row : (M - 1);
                As[m][kk] = f2bf(srcp[(size_t)row * lds_ + koff + kk]);
            }
        } else {
            #pragma unroll
            for (int i = 0; i < (BM * BK) / 256; i++) {
                int linear = t + i * 256;
                int kk  = linear & (BK - 1);
                int m   = linear >> 5;
                int row = rowTile + m;
                row = (row < M) ? row : (M - 1);
                int gk  = k0 + kk;
                int gkc = (gk < Ktot - 1) ? gk : (Ktot - 1);
                const float* p = (gkc < K0)
                    ? (A0 + (size_t)row * lda0 + gkc)
                    : (A1 + (size_t)row * lda1 + (gkc - K0));  // branchless addr select
                float lv = *p;
                As[m][kk] = f2bf((gk < Ktot) ? lv : 0.0f);
            }
        }

        if (k0 + BK < Ktot)
            __builtin_prefetch(Wt + (size_t)(colTile + srow) * Kpad + k0 + BK, 0, 1);

#if HAVE_TDM
        if (wave == 0) __builtin_amdgcn_s_wait_tensorcnt(0);  // TDM done before barrier
#endif
        __syncthreads();

        // ---- fragments per ISA 16-bit wave32 layouts ----
        FragBF af[4];
        #pragma unroll
        for (int mt = 0; mt < 4; mt++) {
            const unsigned short* ar = &As[wm * 64 + mt * 16 + lo][0];
            #pragma unroll
            for (int e = 0; e < 8; e++) af[mt].u[e]     = ar[hi * 8 + e];
            #pragma unroll
            for (int e = 0; e < 8; e++) af[mt].u[8 + e] = ar[16 + hi * 8 + e];
        }
        FragBF bfr[2];
        #pragma unroll
        for (int nt = 0; nt < 2; nt++) {
            const unsigned short* br = &Bs[wn * 32 + nt * 16 + lo][0];
            #pragma unroll
            for (int e = 0; e < 8; e++) bfr[nt].u[e]     = br[hi * 8 + e];
            #pragma unroll
            for (int e = 0; e < 8; e++) bfr[nt].u[8 + e] = br[16 + hi * 8 + e];
        }

        #pragma unroll
        for (int mt = 0; mt < 4; mt++)
            #pragma unroll
            for (int nt = 0; nt < 2; nt++)
                acc[mt][nt] = __builtin_amdgcn_wmma_f32_16x16x32_bf16(
                    false, af[mt].v, false, bfr[nt].v,
                    (short)0, acc[mt][nt], false, false);

        __syncthreads();
    }

    // ---- epilogue: Base + bias + relu, optional pre-activation store ----
    #pragma unroll
    for (int mt = 0; mt < 4; mt++) {
        int rowBase = rowTile + wm * 64 + mt * 16 + hi * 8;  // C/D layout
        #pragma unroll
        for (int nt = 0; nt < 2; nt++) {
            int col = colTile + wn * 32 + nt * 16 + lo;
            #pragma unroll
            for (int r = 0; r < 8; r++) {
                int row = rowBase + r;
                if (row < M) {
                    float v = acc[mt][nt][r];
                    if (Base) v += Base[(size_t)row * N + col];
                    if (bias) v += bias[col];
                    if (Cpre) Cpre[(size_t)row * N + col] = v;
                    C[(size_t)row * N + col] = doRelu ? fmaxf(v, 0.0f) : v;
                }
            }
        }
    }
}

// ---------------------------------------------------------------------------
// Gather / reduction kernels (bandwidth-bound, float4 vectorized, H = 256).
// ---------------------------------------------------------------------------

__global__ void __launch_bounds__(256)
atom_aggregate(const float* __restrict__ message, const int* __restrict__ a2b,
               float* __restrict__ amsg, int n_atoms)
{
    int idx  = blockIdx.x * blockDim.x + threadIdx.x;
    int atom = idx >> 6;
    int c    = (idx & 63) << 2;
    if (atom >= n_atoms) return;
    float4 s = make_float4(0.f, 0.f, 0.f, 0.f);
    #pragma unroll
    for (int k = 0; k < 6; k++) {
        int b = a2b[atom * 6 + k];
        const float4 v = *reinterpret_cast<const float4*>(message + (size_t)b * 256 + c);
        s.x += v.x; s.y += v.y; s.z += v.z; s.w += v.w;
    }
    *reinterpret_cast<float4*>(amsg + (size_t)atom * 256 + c) = s;
}

__global__ void __launch_bounds__(256)
bond_delta(const float* __restrict__ amsg, const float* __restrict__ message,
           const int* __restrict__ b2a, const int* __restrict__ b2revb,
           float* __restrict__ delta, int n_bonds)
{
    int idx  = blockIdx.x * blockDim.x + threadIdx.x;
    int bond = idx >> 6;
    int c    = (idx & 63) << 2;
    if (bond >= n_bonds) return;
    int a  = b2a[bond];
    int rb = b2revb[bond];
    const float4 x = *reinterpret_cast<const float4*>(amsg + (size_t)a * 256 + c);
    const float4 y = *reinterpret_cast<const float4*>(message + (size_t)rb * 256 + c);
    float4 d = make_float4(x.x - y.x, x.y - y.y, x.z - y.z, x.w - y.w);
    *reinterpret_cast<float4*>(delta + (size_t)bond * 256 + c) = d;
}

__global__ void __launch_bounds__(256)
zero_f32(float* __restrict__ p, size_t n)
{
    size_t i = (size_t)blockIdx.x * blockDim.x + threadIdx.x;
    if (i < n) p[i] = 0.0f;
}

__global__ void __launch_bounds__(256)
segsum(const float* __restrict__ ah, const int* __restrict__ atom2mol,
       float* __restrict__ sums, float* __restrict__ counts, int n_atoms)
{
    int idx  = blockIdx.x * blockDim.x + threadIdx.x;
    int atom = idx >> 6;
    int c    = (idx & 63) << 2;
    if (atom >= n_atoms) return;
    int m = atom2mol[atom];
    const float4 v = *reinterpret_cast<const float4*>(ah + (size_t)atom * 256 + c);
    atomicAdd(&sums[(size_t)m * 256 + c + 0], v.x);
    atomicAdd(&sums[(size_t)m * 256 + c + 1], v.y);
    atomicAdd(&sums[(size_t)m * 256 + c + 2], v.z);
    atomicAdd(&sums[(size_t)m * 256 + c + 3], v.w);
    if (c == 0) atomicAdd(&counts[m], 1.0f);
}

__global__ void __launch_bounds__(256)
segdiv(const float* __restrict__ sums, const float* __restrict__ counts,
       float* __restrict__ out, int n_mols)
{
    int idx = blockIdx.x * blockDim.x + threadIdx.x;
    int m = idx >> 8;
    int c = idx & 255;
    if (m >= n_mols) return;
    float cnt = fmaxf(counts[m], 1.0f);
    out[(size_t)m * 256 + c] = sums[(size_t)m * 256 + c] / cnt;
}

// ---------------------------------------------------------------------------
// Host orchestration
// ---------------------------------------------------------------------------

extern "C" void kernel_launch(void* const* d_in, const int* in_sizes, int n_in,
                              void* d_out, int out_size, void* d_ws, size_t ws_size,
                              hipStream_t stream)
{
    const float* f_atoms  = (const float*)d_in[0];
    const float* f_bonds  = (const float*)d_in[1];
    const int*   a2b      = (const int*)d_in[2];
    const int*   b2a      = (const int*)d_in[3];
    const int*   b2revb   = (const int*)d_in[4];
    const int*   atom2mol = (const int*)d_in[5];
    const float* mol_feat = (const float*)d_in[6];
    const float* W_i      = (const float*)d_in[7];
    const float* W_h      = (const float*)d_in[8];
    const float* W_o_w    = (const float*)d_in[9];
    const float* W_o_b    = (const float*)d_in[10];
    const float* W1       = (const float*)d_in[11];
    const float* b1       = (const float*)d_in[12];
    const float* W2       = (const float*)d_in[13];
    const float* b2       = (const float*)d_in[14];
    (void)in_sizes; (void)n_in; (void)out_size; (void)ws_size;

    const int N_ATOMS = 100000, N_BONDS = 200000;
    const int ATOM_FDIM = 133, BOND_FDIM = 147;
    const int H = 256, NMOLS = 4096, MOLF = 200, FFNN = 512, ENC = 256;

    // K paddings (multiple of BK=32)
    const int KP_I = 160;   // 147
    const int KP_H = 256;   // 256
    const int KP_O = 416;   // 389
    const int KP_1 = 480;   // 456
    const int KP_2 = 512;   // 512

    float* ws = (float*)d_ws;
    size_t off = 0;
    float* inputs  = ws + off; off += (size_t)N_BONDS * H;   // pre-activation
    float* message = ws + off; off += (size_t)N_BONDS * H;
    float* amsg    = ws + off; off += (size_t)N_ATOMS * H;
    float* delta   = ws + off; off += (size_t)N_BONDS * H;   // reused: atom_hiddens
    float* molsum  = ws + off; off += (size_t)NMOLS * H;
    float* counts  = ws + off; off += (size_t)NMOLS;
    float* molvec  = ws + off; off += (size_t)NMOLS * H;
    float* hbuf    = ws + off; off += (size_t)NMOLS * FFNN;

    unsigned short* wt = (unsigned short*)(ws + off);
    size_t uoff = 0;
    unsigned short* wt_i = wt + uoff; uoff += (size_t)H    * KP_I;
    unsigned short* wt_h = wt + uoff; uoff += (size_t)H    * KP_H;
    unsigned short* wt_o = wt + uoff; uoff += (size_t)H    * KP_O;
    unsigned short* wt_1 = wt + uoff; uoff += (size_t)FFNN * KP_1;
    unsigned short* wt_2 = wt + uoff; uoff += (size_t)ENC  * KP_2;

    dim3 blk(256);

    // 0) pack all weights to transposed bf16 (tiny, one pass)
    pack_wt<<<dim3((KP_I + 255) / 256, H),    blk, 0, stream>>>(W_i,   wt_i, BOND_FDIM,     H,    KP_I);
    pack_wt<<<dim3((KP_H + 255) / 256, H),    blk, 0, stream>>>(W_h,   wt_h, H,             H,    KP_H);
    pack_wt<<<dim3((KP_O + 255) / 256, H),    blk, 0, stream>>>(W_o_w, wt_o, ATOM_FDIM + H, H,    KP_O);
    pack_wt<<<dim3((KP_1 + 255) / 256, FFNN), blk, 0, stream>>>(W1,    wt_1, H + MOLF,      FFNN, KP_1);
    pack_wt<<<dim3((KP_2 + 255) / 256, ENC),  blk, 0, stream>>>(W2,    wt_2, FFNN,          ENC,  KP_2);

    // 1) inputs = f_bonds @ W_i ; message = relu(inputs)
    {
        dim3 grid(H / BN, (N_BONDS + BM - 1) / BM);
        gemm_bf16_wmma<<<grid, blk, 0, stream>>>(f_bonds, BOND_FDIM, BOND_FDIM,
            nullptr, 0, 0, wt_i, KP_I, nullptr, nullptr, inputs, message, N_BONDS, H, 1);
    }

    // 2) DEPTH-1 = 2 message-passing iterations
    for (int d = 0; d < 2; d++) {
        atom_aggregate<<<(N_ATOMS * 64 + 255) / 256, blk, 0, stream>>>(message, a2b, amsg, N_ATOMS);
        bond_delta<<<(N_BONDS * 64 + 255) / 256, blk, 0, stream>>>(amsg, message, b2a, b2revb, delta, N_BONDS);
        dim3 grid(H / BN, (N_BONDS + BM - 1) / BM);
        gemm_bf16_wmma<<<grid, blk, 0, stream>>>(delta, H, H, nullptr, 0, 0,
            wt_h, KP_H, inputs, nullptr, nullptr, message, N_BONDS, H, 1);
    }

    // 3) final aggregation
    atom_aggregate<<<(N_ATOMS * 64 + 255) / 256, blk, 0, stream>>>(message, a2b, amsg, N_ATOMS);

    // 4) atom_hiddens = relu([f_atoms | amsg] @ W_o + b)  (split-K, no concat buffer)
    float* atom_hiddens = delta;
    {
        dim3 grid(H / BN, (N_ATOMS + BM - 1) / BM);
        gemm_bf16_wmma<<<grid, blk, 0, stream>>>(f_atoms, ATOM_FDIM, ATOM_FDIM,
            amsg, H, H, wt_o, KP_O, nullptr, W_o_b, nullptr, atom_hiddens, N_ATOMS, H, 1);
    }

    // 5) per-molecule mean (molsum and counts contiguous -> one zero pass)
    {
        size_t nz = (size_t)NMOLS * H + NMOLS;
        zero_f32<<<(unsigned)((nz + 255) / 256), blk, 0, stream>>>(molsum, nz);
        segsum<<<(N_ATOMS * 64 + 255) / 256, blk, 0, stream>>>(atom_hiddens, atom2mol, molsum, counts, N_ATOMS);
        segdiv<<<((NMOLS * H) + 255) / 256, blk, 0, stream>>>(molsum, counts, molvec, NMOLS);
    }

    // 6) h = relu([molvec | mol_feat] @ W1 + b1)
    {
        dim3 grid(FFNN / BN, (NMOLS + BM - 1) / BM);
        gemm_bf16_wmma<<<grid, blk, 0, stream>>>(molvec, H, H, mol_feat, MOLF, MOLF,
            wt_1, KP_1, nullptr, b1, nullptr, hbuf, NMOLS, FFNN, 1);
    }

    // 7) out = relu(h @ W2 + b2)
    {
        dim3 grid(ENC / BN, (NMOLS + BM - 1) / BM);
        gemm_bf16_wmma<<<grid, blk, 0, stream>>>(hbuf, FFNN, FFNN, nullptr, 0, 0,
            wt_2, KP_2, nullptr, b2, nullptr, (float*)d_out, NMOLS, ENC, 1);
    }
}
